// MatchNet_21646635172526
// MI455X (gfx1250) — matrix operand order repaired and checked by reading, not compiled, not run on hardware
//
#include <hip/hip_runtime.h>
#include <hip/hip_bf16.h>

#define NN 50000
#define MM 128
#define EE 800000
#define BB 100000

typedef __attribute__((ext_vector_type(16))) __bf16 v16bf;
typedef __attribute__((ext_vector_type(8)))  float  v8f;

union Frag {
    v16bf v;
    unsigned short u[16];
    uint4 q[2];          // two 16B halves: K=kb8+0..7 and K=kb8+16..23
};

// float -> bf16 round-to-nearest-even (bit pattern)
static __device__ __forceinline__ unsigned short f2bf(float x) {
    unsigned u = __float_as_uint(x);
    unsigned r = u + 0x7FFFu + ((u >> 16) & 1u);
    return (unsigned short)(r >> 16);
}

static __device__ __forceinline__ void wave_lds_fence() {
    // cross-lane LDS dependency within a wave: compiler barrier + wait DS ops
    asm volatile("s_wait_dscnt 0x0" ::: "memory");
}

#define NW 8   // waves per block (256 threads, wave32)

// Streaming pass: bf = bf16(src); optionally cpy = src (seeds the f32 accumulator).
__global__ __launch_bounds__(256) void cvt_pass(
    const float* __restrict__ src, unsigned short* __restrict__ bf,
    float* __restrict__ cpy, int n4)   // n4 = (N*M)/4
{
    int i = blockIdx.x * blockDim.x + threadIdx.x;
    const int stride = gridDim.x * blockDim.x;
    const float4* __restrict__ s4 = (const float4*)src;
    for (; i < n4; i += stride) {
        float4 x = s4[i];
        if (cpy) ((float4*)cpy)[i] = x;
        ushort4 b;
        b.x = f2bf(x.x); b.y = f2bf(x.y); b.z = f2bf(x.z); b.w = f2bf(x.w);
        ((ushort4*)bf)[i] = b;
    }
}

// One message-passing step: nxt[dst] += MLP(concat(score[src],score[dst]))
// scoreBf is the bf16 mirror of the current score; nxt is pre-seeded with score.
__global__ __launch_bounds__(256) void edge_pass(
    const unsigned short* __restrict__ scoreBf, float* __restrict__ nxt,
    const long long* __restrict__ edges,
    const float* __restrict__ w1, const float* __restrict__ b1,
    const float* __restrict__ w2, const float* __restrict__ b2,
    const float* __restrict__ w3, const float* __restrict__ b3,
    int numTiles)
{
    __shared__ unsigned short w1p[256 * 16];   // 256 x 16 bf16 (7 cols valid)
    __shared__ unsigned short w2p[32 * 16];    // 32 x 16 bf16 (7x7 valid)
    __shared__ unsigned short w3p[32 * 128];   // 32 x 128 bf16 (7 rows valid)
    __shared__ float b1s[16], b2s[16], b3s[128];
    __shared__ unsigned short htile[NW][16 * 32]; // per-wave h round-trip tile
    __shared__ int dsti[NW][16];

    const int tid = threadIdx.x;

    // --- cooperative weight prep (f32 -> zero-padded bf16) ---
    for (int i = tid; i < 256 * 16; i += blockDim.x) {
        int k = i >> 4, j = i & 15;
        w1p[i] = (j < 7) ? f2bf(w1[k * 7 + j]) : (unsigned short)0;
    }
    for (int i = tid; i < 32 * 16; i += blockDim.x) {
        int k = i >> 4, j = i & 15;
        w2p[i] = (k < 7 && j < 7) ? f2bf(w2[k * 7 + j]) : (unsigned short)0;
    }
    for (int i = tid; i < 32 * 128; i += blockDim.x) {
        int k = i >> 7, j = i & 127;
        w3p[i] = (k < 7) ? f2bf(w3[k * 128 + j]) : (unsigned short)0;
    }
    if (tid < 16) {
        b1s[tid] = (tid < 7) ? b1[tid] : 0.f;
        b2s[tid] = (tid < 7) ? b2[tid] : 0.f;
    }
    for (int i = tid; i < 128; i += blockDim.x) b3s[i] = b3[i];
    for (int i = tid; i < NW * 16 * 32; i += blockDim.x)
        (&htile[0][0])[i] = 0;   // zero pad cols (16..31 stay zero forever)
    __syncthreads();

    const int lane = tid & 31;
    const int wv   = tid >> 5;
    const int col  = lane & 15;          // B/C/D column for this lane
    const int hi   = lane >> 4;          // lane half
    const int kb16 = hi * 16;            // unused (B fragments use kb8 too)
    const int kb8  = hi * 8;             // A-fragment K base

    // --- hoist per-wave B-matrix fragments into registers ---
    Frag w1f[8];
#pragma unroll
    for (int c = 0; c < 8; ++c)
#pragma unroll
        for (int i = 0; i < 16; ++i)
            w1f[c].u[i] = w1p[(32 * c + kb8 + (i < 8 ? i : i + 8)) * 16 + col];

    Frag w2f;
#pragma unroll
    for (int i = 0; i < 16; ++i)
        w2f.u[i] = w2p[(kb8 + (i < 8 ? i : i + 8)) * 16 + col];

    Frag w3f[8];
#pragma unroll
    for (int nt = 0; nt < 8; ++nt)
#pragma unroll
        for (int i = 0; i < 16; ++i)
            w3f[nt].u[i] = w3p[(kb8 + (i < 8 ? i : i + 8)) * 128 + 16 * nt + col];

    const float bias1 = b1s[col];
    const float bias2 = b2s[col];
    const int row = lane & 15;           // edge row handled by this lane (A side)

    const int waveId   = blockIdx.x * NW + wv;
    const int waveStep = gridDim.x * NW;

    for (int t = waveId; t < numTiles; t += waveStep) {
        long long e = (long long)t * 16 + row;
        long long s = edges[2 * e];
        long long d = edges[2 * e + 1];
        if (lane < 16) dsti[wv][row] = (int)d;

        const unsigned short* __restrict__ rs = scoreBf + (size_t)s * MM;
        const unsigned short* __restrict__ rd = scoreBf + (size_t)d * MM;

        // ---- layer 1: (16x256) @ (256x16) via 8 bf16 WMMAs ----
        // batch all A-fragment loads (16 x b128) before the WMMA chain
        Frag a[8];
#pragma unroll
        for (int c = 0; c < 8; ++c) {
            int f0 = c * 32 + kb8;       // 8 consecutive bf16 features (16B)
            int f1 = f0 + 16;            // 8 consecutive bf16 features at f0 + 16
            const unsigned short* p0 = (f0 < MM) ? (rs + f0) : (rd + (f0 - MM));
            const unsigned short* p1 = (f1 < MM) ? (rs + f1) : (rd + (f1 - MM));
            a[c].q[0] = *(const uint4*)p0;
            a[c].q[1] = *(const uint4*)p1;
        }
        v8f c1 = {};
#pragma unroll
        for (int c = 0; c < 8; ++c)
            c1 = __builtin_amdgcn_wmma_f32_16x16x32_bf16(
                     false, a[c].v, false, w1f[c].v, (short)0, c1, false, false);

        // bias + relu, store h1 (bf16) into per-wave tile
#pragma unroll
        for (int r = 0; r < 8; ++r) {
            float v = c1[r] + bias1;
            v = v > 0.f ? v : 0.f;
            htile[wv][(r + 8 * hi) * 32 + col] = f2bf(v);
        }
        wave_lds_fence();

        // ---- layer 2: (16x32 zero-padded) @ (32x16) : 1 WMMA ----
        Frag a2;
#pragma unroll
        for (int i = 0; i < 8; ++i) {
            a2.u[i]     = htile[wv][row * 32 + kb8 + i];
            a2.u[8 + i] = htile[wv][row * 32 + kb8 + 16 + i];
        }
        v8f c2 = {};
        c2 = __builtin_amdgcn_wmma_f32_16x16x32_bf16(
                 false, a2.v, false, w2f.v, (short)0, c2, false, false);
#pragma unroll
        for (int r = 0; r < 8; ++r) {
            float v = c2[r] + bias2;
            v = v > 0.f ? v : 0.f;
            htile[wv][(r + 8 * hi) * 32 + col] = f2bf(v);
        }
        wave_lds_fence();

        // ---- layer 3: (16x32) @ (32x128) as 8 N-tiles, scatter-add ----
        Frag a3;
#pragma unroll
        for (int i = 0; i < 8; ++i) {
            a3.u[i]     = htile[wv][row * 32 + kb8 + i];
            a3.u[8 + i] = htile[wv][row * 32 + kb8 + 16 + i];
        }
        int drows[8];
#pragma unroll
        for (int r = 0; r < 8; ++r) drows[r] = dsti[wv][r + 8 * hi];

#pragma unroll
        for (int nt = 0; nt < 8; ++nt) {
            v8f c3 = {};
            c3 = __builtin_amdgcn_wmma_f32_16x16x32_bf16(
                     false, a3.v, false, w3f[nt].v, (short)0, c3, false, false);
            int gcol = nt * 16 + col;
            float bb = b3s[gcol];
#pragma unroll
            for (int r = 0; r < 8; ++r) {
                unsafeAtomicAdd(&nxt[(size_t)drows[r] * MM + gcol], c3[r] + bb);
            }
        }
    }
}

// Final readout: sigmoid(MLP(concat(score[i0],score[i1])))
__global__ __launch_bounds__(256) void label_pass(
    const unsigned short* __restrict__ scoreBf,
    const long long* __restrict__ lidx,
    const float* __restrict__ w1, const float* __restrict__ b1,
    const float* __restrict__ w2, const float* __restrict__ b2,
    const float* __restrict__ w3, const float* __restrict__ b3,
    float* __restrict__ out, int numTiles)
{
    __shared__ unsigned short w1p[256 * 16];   // 256 x 16 (9 cols valid)
    __shared__ unsigned short w2p[32 * 16];    // 32 x 16 (9x9 valid)
    __shared__ float b1s[16], b2s[16], w3s[16];
    __shared__ float bb3;
    __shared__ unsigned short htile[NW][16 * 32];
    __shared__ float ftile[NW][16 * 16];

    const int tid = threadIdx.x;
    for (int i = tid; i < 256 * 16; i += blockDim.x) {
        int k = i >> 4, j = i & 15;
        w1p[i] = (j < 9) ? f2bf(w1[k * 9 + j]) : (unsigned short)0;
    }
    for (int i = tid; i < 32 * 16; i += blockDim.x) {
        int k = i >> 4, j = i & 15;
        w2p[i] = (k < 9 && j < 9) ? f2bf(w2[k * 9 + j]) : (unsigned short)0;
    }
    if (tid < 16) {
        b1s[tid] = (tid < 9) ? b1[tid] : 0.f;
        b2s[tid] = (tid < 9) ? b2[tid] : 0.f;
        w3s[tid] = (tid < 9) ? w3[tid] : 0.f;
    }
    if (tid == 0) bb3 = b3[0];
    for (int i = tid; i < NW * 16 * 32; i += blockDim.x)
        (&htile[0][0])[i] = 0;
    __syncthreads();

    const int lane = tid & 31;
    const int wv   = tid >> 5;
    const int col  = lane & 15;
    const int hi   = lane >> 4;
    const int kb16 = hi * 16;
    const int kb8  = hi * 8;
    const int row  = lane & 15;

    Frag w1f[8];
#pragma unroll
    for (int c = 0; c < 8; ++c)
#pragma unroll
        for (int i = 0; i < 16; ++i)
            w1f[c].u[i] = w1p[(32 * c + kb8 + (i < 8 ? i : i + 8)) * 16 + col];
    Frag w2f;
#pragma unroll
    for (int i = 0; i < 16; ++i)
        w2f.u[i] = w2p[(kb8 + (i < 8 ? i : i + 8)) * 16 + col];

    const float bias1 = b1s[col];
    const float bias2 = b2s[col];

    const int waveId   = blockIdx.x * NW + wv;
    const int waveStep = gridDim.x * NW;

    for (int t = waveId; t < numTiles; t += waveStep) {
        long long e = (long long)t * 16 + row;
        long long i0 = lidx[2 * e];
        long long i1 = lidx[2 * e + 1];
        const unsigned short* __restrict__ r0 = scoreBf + (size_t)i0 * MM;
        const unsigned short* __restrict__ r1 = scoreBf + (size_t)i1 * MM;

        Frag a[8];
#pragma unroll
        for (int c = 0; c < 8; ++c) {
            int f0 = c * 32 + kb8;
            int f1 = f0 + 16;
            const unsigned short* p0 = (f0 < MM) ? (r0 + f0) : (r1 + (f0 - MM));
            const unsigned short* p1 = (f1 < MM) ? (r0 + f1) : (r1 + (f1 - MM));
            a[c].q[0] = *(const uint4*)p0;
            a[c].q[1] = *(const uint4*)p1;
        }
        v8f c1 = {};
#pragma unroll
        for (int c = 0; c < 8; ++c)
            c1 = __builtin_amdgcn_wmma_f32_16x16x32_bf16(
                     false, a[c].v, false, w1f[c].v, (short)0, c1, false, false);

#pragma unroll
        for (int r = 0; r < 8; ++r) {
            float v = c1[r] + bias1;
            v = v > 0.f ? v : 0.f;
            htile[wv][(r + 8 * hi) * 32 + col] = f2bf(v);
        }
        wave_lds_fence();

        Frag a2;
#pragma unroll
        for (int i = 0; i < 8; ++i) {
            a2.u[i]     = htile[wv][row * 32 + kb8 + i];
            a2.u[8 + i] = htile[wv][row * 32 + kb8 + 16 + i];
        }
        v8f c2 = {};
        c2 = __builtin_amdgcn_wmma_f32_16x16x32_bf16(
                 false, a2.v, false, w2f.v, (short)0, c2, false, false);
#pragma unroll
        for (int r = 0; r < 8; ++r) {
            float v = c2[r] + bias2;
            ftile[wv][(r + 8 * hi) * 16 + col] = v > 0.f ? v : 0.f;
        }
        wave_lds_fence();

        if (lane < 16) {
            float logit = bb3;
#pragma unroll
            for (int k = 0; k < 9; ++k)
                logit += ftile[wv][row * 16 + k] * w3s[k];
            out[(size_t)t * 16 + row] = 1.f / (1.f + __expf(-logit));
        }
        wave_lds_fence();   // protect ftile WAR across loop iterations
    }
}

extern "C" void kernel_launch(void* const* d_in, const int* in_sizes, int n_in,
                              void* d_out, int out_size, void* d_ws, size_t ws_size,
                              hipStream_t stream) {
    const float*     score = (const float*)d_in[0];
    const long long* edges = (const long long*)d_in[1];
    const long long* lidx  = (const long long*)d_in[2];
    const float* rp_w1 = (const float*)d_in[3];
    const float* rp_b1 = (const float*)d_in[4];
    const float* rp_w2 = (const float*)d_in[5];
    const float* rp_b2 = (const float*)d_in[6];
    const float* rp_w3 = (const float*)d_in[7];
    const float* rp_b3 = (const float*)d_in[8];
    const float* g_w1  = (const float*)d_in[9];
    const float* g_b1  = (const float*)d_in[10];
    const float* g_w2  = (const float*)d_in[11];
    const float* g_b2  = (const float*)d_in[12];
    const float* g_w3  = (const float*)d_in[13];
    const float* g_b3  = (const float*)d_in[14];
    float* out = (float*)d_out;

    // workspace layout: s0 (f32), s1 (f32), sb (bf16 mirror)
    float* s0 = (float*)d_ws;
    float* s1 = s0 + (size_t)NN * MM;
    unsigned short* sb = (unsigned short*)(s1 + (size_t)NN * MM);

    const int n4 = (NN * MM) / 4;     // 1.6M float4 groups
    const int edgeTiles  = EE / 16;   // 50000
    const int labelTiles = BB / 16;   // 6250
    const dim3 cg(2048), eg(640), lb(512), th(256);

    // step 1: sb = bf16(score), s0 = score; s0 += agg
    cvt_pass<<<cg, th, 0, stream>>>(score, sb, s0, n4);
    edge_pass<<<eg, th, 0, stream>>>(sb, s0, edges,
        rp_w1, rp_b1, rp_w2, rp_b2, rp_w3, rp_b3, edgeTiles);
    // step 2: sb = bf16(s0), s1 = s0; s1 += agg
    cvt_pass<<<cg, th, 0, stream>>>(s0, sb, s1, n4);
    edge_pass<<<eg, th, 0, stream>>>(sb, s1, edges,
        rp_w1, rp_b1, rp_w2, rp_b2, rp_w3, rp_b3, edgeTiles);
    // step 3: sb = bf16(s1), s0 = s1; s0 += agg
    cvt_pass<<<cg, th, 0, stream>>>(s1, sb, s0, n4);
    edge_pass<<<eg, th, 0, stream>>>(sb, s0, edges,
        rp_w1, rp_b1, rp_w2, rp_b2, rp_w3, rp_b3, edgeTiles);
    // readout: sb = bf16(s0), no f32 copy needed
    cvt_pass<<<cg, th, 0, stream>>>(s0, sb, nullptr, n4);
    label_pass<<<lb, th, 0, stream>>>(sb, lidx,
        g_w1, g_b1, g_w2, g_b2, g_w3, g_b3, out, labelTiles);
}
